// GATTox21_8693013807140
// MI455X (gfx1250) — matrix operand order repaired and checked by reading, not compiled, not run on hardware
//
#include <hip/hip_runtime.h>
#include <hip/hip_bf16.h>

typedef __bf16 bf16;
typedef __attribute__((ext_vector_type(16))) __bf16 v16bf;
typedef __attribute__((ext_vector_type(8)))  __bf16 v8bf;
typedef __attribute__((ext_vector_type(8)))  float  v8f;

#define N_NODES 20000
#define N_EDGES 320000
#define E2_EDGES (N_EDGES + N_NODES)
#define F_NODE 32
#define F_EDGE 16
#define HID 128
#define OUTC 256           // HEADS*HIDDEN
#define NGRAPH 256
#define NTASK 12
#define NEG_SLOPE 0.2f
#define BN_EPS 1e-5f

// ---------------- small utility kernels ----------------

__global__ void zero_f32(float* p, int n) {
    int i = blockIdx.x * blockDim.x + threadIdx.x;
    if (i < n) p[i] = 0.0f;
}

__global__ void f2bf_kernel(const float* __restrict__ in, bf16* __restrict__ out, int n) {
    int i = blockIdx.x * blockDim.x + threadIdx.x;
    if (i < n) out[i] = (bf16)in[i];
}

// in: K x Nc (row major f32) -> out: Nc x K (row major bf16)
__global__ void transpose_f2bf_kernel(const float* __restrict__ in, bf16* __restrict__ out,
                                      int K, int Nc) {
    int i = blockIdx.x * blockDim.x + threadIdx.x;
    if (i < K * Nc) {
        int k = i / Nc;
        int n = i - k * Nc;
        out[n * K + k] = (bf16)in[i];
    }
}

__global__ void ea_sum_kernel(const float* __restrict__ edge_attr, float* __restrict__ ea_acc,
                              int total) {
    __shared__ float s[F_EDGE];
    if (threadIdx.x < F_EDGE) s[threadIdx.x] = 0.0f;
    __syncthreads();
    for (int i = blockIdx.x * blockDim.x + threadIdx.x; i < total; i += gridDim.x * blockDim.x)
        atomicAdd(&s[i & (F_EDGE - 1)], edge_attr[i]);
    __syncthreads();
    if (threadIdx.x < F_EDGE) atomicAdd(&ea_acc[threadIdx.x], s[threadIdx.x]);
}

// ---------------- WMMA GEMM: C[M x Nc] = A[M x K](bf16) * B (Bt stored Nc x K bf16) ------------
// One wave computes a 16x64 output block (1 M-tile x 4 N-tiles): the A fragment is loaded once
// per k-step and reused by 4 independent WMMA accumulator chains.
// A and B fragments both use contiguous k order, k = 16*half + s (wave32). Requires Nc % 64 == 0, M % 16 == 0, K % 32 == 0.

__global__ void gemm_bf16_wmma(const bf16* __restrict__ A, const bf16* __restrict__ Bt,
                               float* __restrict__ C, const float* __restrict__ bias,
                               int M, int K, int Nc) {
    int nBlkN  = Nc >> 6;                       // 64-column blocks
    int nWaves = (M >> 4) * nBlkN;
    int wave = (blockIdx.x * blockDim.x + threadIdx.x) >> 5;
    if (wave >= nWaves) return;
    int tm = wave / nBlkN;
    int nb = wave - tm * nBlkN;
    int n0 = nb << 6;
    int lane = threadIdx.x & 31;
    int half = lane >> 4;      // 0: lanes 0-15, 1: lanes 16-31
    int l16  = lane & 15;

    const bf16* Arow = A  + (size_t)(tm * 16 + l16) * K;   // A: row m striped in-lane
    const bf16* Brow = Bt + (size_t)(n0 + l16) * K;        // B: column n contiguous (pre-transposed)
    const size_t bstep = (size_t)16 * K;                   // next 16-column tile

    v8f acc0 = {}, acc1 = {}, acc2 = {}, acc3 = {};
    for (int k0 = 0; k0 < K; k0 += 32) {
        // A 16x32 bf16 fragment: lanes<16 hold K k0..k0+15, lanes>=16 hold K k0+16..k0+31
        v8bf alo = *(const v8bf*)(Arow + k0 + half * 16);
        v8bf ahi = *(const v8bf*)(Arow + k0 + half * 16 + 8);
        v16bf afrag;
#pragma unroll
        for (int j = 0; j < 8; ++j) { afrag[j] = alo[j]; afrag[8 + j] = ahi[j]; }
        // B 32x16 fragments: lanes<16 col n K=k0..k0+15; lanes>=16 col n K=k0+16..k0+31
        v16bf b0 = *(const v16bf*)(Brow + 0 * bstep + k0 + half * 16);
        v16bf b1 = *(const v16bf*)(Brow + 1 * bstep + k0 + half * 16);
        v16bf b2 = *(const v16bf*)(Brow + 2 * bstep + k0 + half * 16);
        v16bf b3 = *(const v16bf*)(Brow + 3 * bstep + k0 + half * 16);
        acc0 = __builtin_amdgcn_wmma_f32_16x16x32_bf16(false, afrag, false, b0, (short)0, acc0, false, false);
        acc1 = __builtin_amdgcn_wmma_f32_16x16x32_bf16(false, afrag, false, b1, (short)0, acc1, false, false);
        acc2 = __builtin_amdgcn_wmma_f32_16x16x32_bf16(false, afrag, false, b2, (short)0, acc2, false, false);
        acc3 = __builtin_amdgcn_wmma_f32_16x16x32_bf16(false, afrag, false, b3, (short)0, acc3, false, false);
    }
    // C/D layout: VGPR r -> row (half*8 + r), col l16 within each 16x16 tile
    float* Crow = C + (size_t)(tm * 16 + half * 8) * Nc + n0 + l16;
    v8f accs[4] = {acc0, acc1, acc2, acc3};
#pragma unroll
    for (int nt = 0; nt < 4; ++nt) {
        int n = n0 + nt * 16 + l16;
        float b = bias ? bias[n] : 0.0f;
        float* Cp = Crow + nt * 16;
#pragma unroll
        for (int r = 0; r < 8; ++r) Cp[(size_t)r * Nc] = accs[nt][r] + b;
    }
}

// ---------------- edge phase ----------------

__device__ __forceinline__ unsigned fkey(float f) {
    unsigned u = __float_as_uint(f);
    return u ^ ((unsigned)((int)u >> 31) | 0x80000000u);
}
__device__ __forceinline__ float funkey(unsigned k) {
    unsigned u = (k & 0x80000000u) ? (k ^ 0x80000000u) : ~k;
    return __uint_as_float(u);
}

// one wave per edge; lane covers 8 consecutive channels of the 256 (=2 heads x 128)
__global__ void edge_logits_kernel(const float* __restrict__ xl, const float* __restrict__ xr,
                                   const int* __restrict__ src, const int* __restrict__ dst,
                                   const float* __restrict__ edge_attr,
                                   const float* __restrict__ ea_acc,
                                   const float* __restrict__ w_e, const float* __restrict__ att,
                                   float* __restrict__ logits, unsigned* __restrict__ lmax,
                                   int E, int E2) {
    __shared__ float s_we[F_EDGE * OUTC];
    __shared__ float s_att[OUTC];
    for (int i = threadIdx.x; i < F_EDGE * OUTC; i += blockDim.x) s_we[i] = w_e[i];
    for (int i = threadIdx.x; i < OUTC; i += blockDim.x) s_att[i] = att[i];
    __syncthreads();

    int wid = (blockIdx.x * blockDim.x + threadIdx.x) >> 5;
    if (wid >= E2) return;
    int lane = threadIdx.x & 31;
    int half = lane >> 4;

    int s, d;
    float ea[F_EDGE];
    if (wid < E) {
        s = src[wid]; d = dst[wid];
#pragma unroll
        for (int k = 0; k < F_EDGE; ++k) ea[k] = edge_attr[wid * F_EDGE + k];
    } else {
        s = wid - E; d = s;
        float inv = 1.0f / (float)E;
#pragma unroll
        for (int k = 0; k < F_EDGE; ++k) ea[k] = ea_acc[k] * inv;
    }

    int f0 = lane * 8;
    const float* xlp = xl + (size_t)s * OUTC + f0;
    const float* xrp = xr + (size_t)d * OUTC + f0;
    float partial = 0.0f;
#pragma unroll
    for (int j = 0; j < 8; ++j) {
        int f = f0 + j;
        float eh = 0.0f;
#pragma unroll
        for (int k = 0; k < F_EDGE; ++k) eh = fmaf(ea[k], s_we[k * OUTC + f], eh);
        float m = xlp[j] + xrp[j] + eh;
        m = (m > 0.0f) ? m : NEG_SLOPE * m;
        partial = fmaf(s_att[f], m, partial);
    }
    // reduce within each half-wave (one head per half)
    partial += __shfl_xor(partial, 1);
    partial += __shfl_xor(partial, 2);
    partial += __shfl_xor(partial, 4);
    partial += __shfl_xor(partial, 8);
    if ((lane & 15) == 0) {
        logits[wid * 2 + half] = partial;
        atomicMax(&lmax[d * 2 + half], fkey(partial));
    }
}

__global__ void edge_exp_kernel(float* __restrict__ logits, const int* __restrict__ dst,
                                const unsigned* __restrict__ lmax, float* __restrict__ den,
                                int E, int E2) {
    int t = blockIdx.x * blockDim.x + threadIdx.x;
    if (t >= E2 * 2) return;
    int e = t >> 1, h = t & 1;
    int d = (e < E) ? dst[e] : (e - E);
    float ex = __expf(logits[t] - funkey(lmax[d * 2 + h]));
    logits[t] = ex;                     // reuse buffer as ex
    atomicAdd(&den[d * 2 + h], ex);
}

__global__ void edge_scatter_kernel(const float* __restrict__ ex, const float* __restrict__ den,
                                    const float* __restrict__ xl,
                                    const int* __restrict__ src, const int* __restrict__ dst,
                                    float* __restrict__ agg, int E, int E2) {
    int wid = (blockIdx.x * blockDim.x + threadIdx.x) >> 5;
    if (wid >= E2) return;
    int lane = threadIdx.x & 31;
    int h = lane >> 4;
    int s, d;
    if (wid < E) { s = src[wid]; d = dst[wid]; }
    else         { s = wid - E; d = s; }
    float alpha = ex[wid * 2 + h] / den[d * 2 + h];
    int f0 = lane * 8;
    const float* xp = xl + (size_t)s * OUTC + f0;
    float* op = agg + (size_t)d * OUTC + f0;
#pragma unroll
    for (int j = 0; j < 8; ++j) atomicAdd(&op[j], alpha * xp[j]);
}

// ---------------- batchnorm + relu ----------------

#define BN_ROWS 128
__global__ void bn_stats_kernel(const float* __restrict__ agg, float* __restrict__ gsum,
                                float* __restrict__ gsq, int N) {
    int c = threadIdx.x;                 // 256 threads = one per column (coalesced rows)
    int r0 = blockIdx.x * BN_ROWS;
    int rend = min(r0 + BN_ROWS, N);
    float s = 0.0f, s2 = 0.0f;
    for (int r = r0; r < rend; ++r) {
        float v = agg[(size_t)r * OUTC + c];
        s += v;
        s2 = fmaf(v, v, s2);
    }
    atomicAdd(&gsum[c], s);
    atomicAdd(&gsq[c], s2);
}

__global__ void bn_apply_kernel(const float* __restrict__ agg, const float* __restrict__ gsum,
                                const float* __restrict__ gsq, const float* __restrict__ gamma,
                                const float* __restrict__ beta, float* __restrict__ h,
                                bf16* __restrict__ hb, int N) {
    int i = blockIdx.x * blockDim.x + threadIdx.x;
    if (i >= N * OUTC) return;
    int c = i & (OUTC - 1);
    float invN = 1.0f / (float)N;
    float m = gsum[c] * invN;
    float var = gsq[c] * invN - m * m;
    float v = gamma[c] * (agg[i] - m) * rsqrtf(var + BN_EPS) + beta[c];
    v = (v > 0.0f) ? v : 0.0f;
    h[i] = v;
    hb[i] = (bf16)v;
}

// ---------------- pooling + head ----------------

__global__ void count_kernel(const int* __restrict__ batch, float* __restrict__ counts, int N) {
    int n = blockIdx.x * blockDim.x + threadIdx.x;
    if (n < N) atomicAdd(&counts[batch[n]], 1.0f);
}

__global__ void pool_kernel(const float* __restrict__ h, const int* __restrict__ batch,
                            float* __restrict__ psum, int N) {
    int i = blockIdx.x * blockDim.x + threadIdx.x;
    if (i < N * OUTC) {
        int n = i >> 8, c = i & (OUTC - 1);
        atomicAdd(&psum[(size_t)batch[n] * OUTC + c], h[i]);
    }
}

__global__ void head_kernel(const float* __restrict__ psum, const float* __restrict__ counts,
                            const float* __restrict__ fcw, const float* __restrict__ fcb,
                            float* __restrict__ out) {
    int g = blockIdx.x;
    int t = threadIdx.x;
    if (t >= NTASK) return;
    float cnt = counts[g];
    cnt = (cnt > 1.0f) ? cnt : 1.0f;
    float inv = 1.0f / cnt;
    float s = fcb[t];
    for (int c = 0; c < OUTC; ++c)
        s = fmaf(psum[g * OUTC + c] * inv, fcw[c * NTASK + t], s);
    out[g * NTASK + t] = 1.0f / (1.0f + __expf(-s));
}

// ---------------- host ----------------

extern "C" void kernel_launch(void* const* d_in, const int* in_sizes, int n_in,
                              void* d_out, int out_size, void* d_ws, size_t ws_size,
                              hipStream_t stream) {
    const int N = N_NODES, E = N_EDGES, E2 = E2_EDGES;

    // -------- input mapping (detect dict-insertion vs pytree-sorted ordering) --------
    const float *x, *edge_attr, *emb_w, *emb_b, *fc_w, *fc_b;
    const int *ei, *batch;
    const float *w_l[4], *w_r[4], *w_e[4], *att[4], *gam[4], *bet[4];
    if (in_sizes[0] == N) {
        // sorted: batch, edge_attr, edge_index, emb_b, emb_w, fc_b, fc_w,
        //         [att,beta,bias,gamma,w_e,w_l,w_r]*4, x
        batch     = (const int*)  d_in[0];
        edge_attr = (const float*)d_in[1];
        ei        = (const int*)  d_in[2];
        emb_b     = (const float*)d_in[3];
        emb_w     = (const float*)d_in[4];
        fc_b      = (const float*)d_in[5];
        fc_w      = (const float*)d_in[6];
        for (int l = 0; l < 4; ++l) {
            int b = 7 + 7 * l;
            att[l] = (const float*)d_in[b + 0];
            bet[l] = (const float*)d_in[b + 1];
            // bias d_in[b+2] unused (cancels through BN)
            gam[l] = (const float*)d_in[b + 3];
            w_e[l] = (const float*)d_in[b + 4];
            w_l[l] = (const float*)d_in[b + 5];
            w_r[l] = (const float*)d_in[b + 6];
        }
        x = (const float*)d_in[35];
    } else {
        // insertion: x, edge_index, edge_attr, batch, emb_w, emb_b,
        //            [w_l,w_r,w_e,att,bias,gamma,beta]*4, fc_w, fc_b
        x         = (const float*)d_in[0];
        ei        = (const int*)  d_in[1];
        edge_attr = (const float*)d_in[2];
        batch     = (const int*)  d_in[3];
        emb_w     = (const float*)d_in[4];
        emb_b     = (const float*)d_in[5];
        for (int l = 0; l < 4; ++l) {
            int b = 6 + 7 * l;
            w_l[l] = (const float*)d_in[b + 0];
            w_r[l] = (const float*)d_in[b + 1];
            w_e[l] = (const float*)d_in[b + 2];
            att[l] = (const float*)d_in[b + 3];
            gam[l] = (const float*)d_in[b + 5];
            bet[l] = (const float*)d_in[b + 6];
        }
        fc_w = (const float*)d_in[34];
        fc_b = (const float*)d_in[35];
    }
    const int* src = ei;
    const int* dst = ei + E;

    // -------- workspace layout --------
    char* base = (char*)d_ws;
    size_t off = 0;
    auto alloc = [&](size_t bytes) -> void* {
        off = (off + 255) & ~((size_t)255);
        void* p = base + off;
        off += bytes;
        return p;
    };
    float*    h      = (float*)   alloc((size_t)N * OUTC * 4);
    bf16*     hb     = (bf16*)    alloc((size_t)N * OUTC * 2);
    bf16*     xb     = (bf16*)    alloc((size_t)N * F_NODE * 2);
    float*    xl     = (float*)   alloc((size_t)N * OUTC * 4);
    float*    xr     = (float*)   alloc((size_t)N * OUTC * 4);
    float*    agg    = (float*)   alloc((size_t)N * OUTC * 4);
    bf16*     wlt    = (bf16*)    alloc((size_t)OUTC * OUTC * 2);
    bf16*     wrt    = (bf16*)    alloc((size_t)OUTC * OUTC * 2);
    bf16*     ewt    = (bf16*)    alloc((size_t)HID * F_NODE * 2);
    float*    logitsb= (float*)   alloc((size_t)E2 * 2 * 4);
    unsigned* lmax   = (unsigned*)alloc((size_t)N * 2 * 4);
    float*    den    = (float*)   alloc((size_t)N * 2 * 4);
    float*    gstats = (float*)   alloc((size_t)2 * OUTC * 4);   // gsum | gsq
    float*    gsum   = gstats;
    float*    gsq    = gstats + OUTC;
    float*    psum   = (float*)   alloc((size_t)NGRAPH * OUTC * 4);
    float*    counts = (float*)   alloc((size_t)NGRAPH * 4);
    float*    eaacc  = (float*)   alloc((size_t)F_EDGE * 4);

    const int TB = 256;
    auto blocks = [](int n, int tb) { return (n + tb - 1) / tb; };

    // -------- edge_attr mean --------
    zero_f32<<<1, 64, 0, stream>>>(eaacc, F_EDGE);
    ea_sum_kernel<<<256, TB, 0, stream>>>(edge_attr, eaacc, E * F_EDGE);

    // -------- node embedding: h = x @ emb_w + emb_b --------
    f2bf_kernel<<<blocks(N * F_NODE, TB), TB, 0, stream>>>(x, xb, N * F_NODE);
    transpose_f2bf_kernel<<<blocks(F_NODE * HID, TB), TB, 0, stream>>>(emb_w, ewt, F_NODE, HID);
    {
        int nWaves = (N / 16) * (HID / 64);              // 2500 waves (16x64 blocks)
        gemm_bf16_wmma<<<blocks(nWaves * 32, TB), TB, 0, stream>>>(xb, ewt, h, emb_b,
                                                                   N, F_NODE, HID);
    }
    f2bf_kernel<<<blocks(N * HID, TB), TB, 0, stream>>>(h, hb, N * HID);

    // -------- GATv2 layers --------
    for (int l = 0; l < 4; ++l) {
        int D = (l == 0) ? HID : OUTC;
        transpose_f2bf_kernel<<<blocks(D * OUTC, TB), TB, 0, stream>>>(w_l[l], wlt, D, OUTC);
        transpose_f2bf_kernel<<<blocks(D * OUTC, TB), TB, 0, stream>>>(w_r[l], wrt, D, OUTC);
        int nWaves = (N / 16) * (OUTC / 64);             // 5000 waves (16x64 blocks)
        gemm_bf16_wmma<<<blocks(nWaves * 32, TB), TB, 0, stream>>>(hb, wlt, xl, nullptr, N, D, OUTC);
        gemm_bf16_wmma<<<blocks(nWaves * 32, TB), TB, 0, stream>>>(hb, wrt, xr, nullptr, N, D, OUTC);

        zero_f32<<<blocks(N * 2, TB), TB, 0, stream>>>((float*)lmax, N * 2);  // 0 == fkey floor
        zero_f32<<<blocks(N * 2, TB), TB, 0, stream>>>(den, N * 2);
        zero_f32<<<blocks(N * OUTC, TB), TB, 0, stream>>>(agg, N * OUTC);

        edge_logits_kernel<<<blocks(E2 * 32, TB), TB, 0, stream>>>(
            xl, xr, src, dst, edge_attr, eaacc, w_e[l], att[l], logitsb, lmax, E, E2);
        edge_exp_kernel<<<blocks(E2 * 2, TB), TB, 0, stream>>>(logitsb, dst, lmax, den, E, E2);
        edge_scatter_kernel<<<blocks(E2 * 32, TB), TB, 0, stream>>>(
            logitsb, den, xl, src, dst, agg, E, E2);

        zero_f32<<<blocks(2 * OUTC, TB), TB, 0, stream>>>(gstats, 2 * OUTC);
        bn_stats_kernel<<<blocks(N, BN_ROWS), OUTC, 0, stream>>>(agg, gsum, gsq, N);
        bn_apply_kernel<<<blocks(N * OUTC, TB), TB, 0, stream>>>(agg, gsum, gsq,
                                                                 gam[l], bet[l], h, hb, N);
    }

    // -------- pooling + head --------
    zero_f32<<<blocks(NGRAPH * OUTC, TB), TB, 0, stream>>>(psum, NGRAPH * OUTC);
    zero_f32<<<1, NGRAPH, 0, stream>>>(counts, NGRAPH);
    count_kernel<<<blocks(N, TB), TB, 0, stream>>>(batch, counts, N);
    pool_kernel<<<blocks(N * OUTC, TB), TB, 0, stream>>>(h, batch, psum, N);
    head_kernel<<<NGRAPH, 16, 0, stream>>>(psum, counts, fc_w, fc_b, (float*)d_out);
}
